// M2BertAttention_81389630259253
// MI455X (gfx1250) — compile-verified, matrix operand order checked by reading, not run on hardware
//
#include <hip/hip_runtime.h>
#include <hip/hip_bf16.h>

typedef __attribute__((ext_vector_type(16))) _Float16 v16h;
typedef __attribute__((ext_vector_type(8)))  _Float16 v8h;
typedef __attribute__((ext_vector_type(2)))  _Float16 v2h;
typedef __attribute__((ext_vector_type(8)))  float    v8f;

#define S_LEN 4096
#define HID   768
#define NH    12
#define HD    64

// ---------------------------------------------------------------------------
// WMMA fragment helpers
// ---------------------------------------------------------------------------
// Load a 16x32 f16 WMMA fragment (A layout; B uses the same loader with the
// tile stored leading-dim-major).  ISA 7.12.2: lane l<16 -> row l, K {0..7,16..23};
// lane l+16 -> row l, K {8..15,24..31}.  ld in halfs, multiple of 8 for 16B align.
__device__ __forceinline__ v16h load_frag16(const _Float16* base, int ld) {
    const int lane = threadIdx.x & 31;
    const int row  = lane & 15;
    const int koff = (lane >> 4) << 3;      // 0 or 8
    const _Float16* p = base + row * ld + koff;
    v8h lo = *(const v8h*)(p);
    v8h hi = *(const v8h*)(p + 16);
    v16h r;
#pragma unroll
    for (int i = 0; i < 8; ++i) { r[i] = lo[i]; r[8 + i] = hi[i]; }
    return r;
}

__device__ __forceinline__ v8f wmma_f16(v16h a, v16h b, v8f c) {
    return __builtin_amdgcn_wmma_f32_16x16x32_f16(false, a, false, b,
                                                  (short)0, c, false, false);
}

// ---------------------------------------------------------------------------
// Tensor Data Mover: 64x64 f16 row-major tile, global -> LDS, with LDS pad so
// the landed tile has stride 72 halfs (pad 16B every 128B row).
// D# layout per CDNA5 ISA 8.3/8.4.
// ---------------------------------------------------------------------------
#if __has_builtin(__builtin_amdgcn_tensor_load_to_lds)
#define HAVE_TDM 1
typedef unsigned int u32x4 __attribute__((ext_vector_type(4)));
typedef int          i32x4 __attribute__((ext_vector_type(4)));
typedef int          i32x8 __attribute__((ext_vector_type(8)));

__device__ __forceinline__ void tdm_load_tile_64x64_f16(const _Float16* gsrc,
                                                        unsigned lds_addr) {
    unsigned long long ga = (unsigned long long)(uintptr_t)gsrc;
    u32x4 g0;
    g0.x = 1u;                                         // count=1, user mode
    g0.y = lds_addr;                                   // [63:32] lds_addr
    g0.z = (unsigned)(ga & 0xFFFFFFFFu);               // [95:64] global lo
    g0.w = (unsigned)((ga >> 32) & 0x1FFFFFFu)         // [120:96] global hi
         | (2u << 30);                                 // [127:126] type=2
    i32x8 g1;
    g1[0] = (1 << 16)            //  data_size = 1 -> 2 bytes
          | (1 << 20)            //  pad_enable
          | (4 << 22)            //  pad_interval: 1<<4 8B-units = 128 B
          | (3 << 25);           //  pad_amount: 4 DWORDs = 16 B
    g1[1] = (int)(64u << 16);    //  tensor_dim0 = 64 (low 16 bits)
    g1[2] = (int)(4096u << 16);  //  tensor_dim0 hi=0 | tensor_dim1 = 4096 lo
    g1[3] = (int)(64u << 16);    //  tensor_dim1 hi=0 | tile_dim0 = 64
    g1[4] = 64;                  //  tile_dim1 = 64, tile_dim2 = 0
    g1[5] = 64;                  //  tensor_dim0_stride = 64 (48-bit, lo dword)
    g1[6] = 0;                   //  stride hi | tensor_dim1_stride lo
    g1[7] = 0;
    i32x4 z4 = {};
#if __has_include(<hip/amd_detail/amd_gfx1250_TDM.h>)
    i32x8 z8 = {};
    __builtin_amdgcn_tensor_load_to_lds(g0, g1, z4, z4, z8, 0);
#else
    __builtin_amdgcn_tensor_load_to_lds(g0, g1, z4, z4, 0);
#endif
}
#endif

// ---------------------------------------------------------------------------
// Kernel 1: fused QKV projection + bias + RoPE, f32 in -> f16 head-major out.
// Block = 256 threads (8 waves), 128(M) x 64(N) tile; N tile == one head so
// RoPE's d <-> d+32 pairing stays in-block.  Global loads for slab k+1 are
// issued into registers while slab k's WMMAs run.
// ---------------------------------------------------------------------------
__global__ __launch_bounds__(256)
void qkv_rope_kernel(const float* __restrict__ hs,
                     const float* __restrict__ Wq, const float* __restrict__ Wk,
                     const float* __restrict__ Wv,
                     const float* __restrict__ bq, const float* __restrict__ bk,
                     const float* __restrict__ bv,
                     _Float16* __restrict__ Qh, _Float16* __restrict__ Kh,
                     _Float16* __restrict__ Vh)
{
    __shared__ __align__(16) _Float16 Ah[128 * 48];   // 128x32 f16, stride 48
    __shared__ __align__(16) _Float16 Bh[64 * 48];    // 64x32  f16, stride 48
    __shared__ __align__(16) float    Cs[128 * 64];   // f32 staging for RoPE

    const int tid  = threadIdx.x;
    const int wave = tid >> 5;
    const int wm   = wave & 3;
    const int wn   = wave >> 2;
    const int m0   = blockIdx.x * 128;
    const int head = blockIdx.y;
    const int n0   = head * HD;
    const int qkv  = blockIdx.z;

    const float* W   = (qkv == 0) ? Wq : (qkv == 1) ? Wk : Wv;
    const float* bia = (qkv == 0) ? bq : (qkv == 1) ? bk : bv;
    _Float16*    dst = (qkv == 0) ? Qh : (qkv == 1) ? Kh : Vh;

    v8f c00 = {}, c01 = {}, c10 = {}, c11 = {};

    // register staging for one k-slab
    float4 ar[4], br[2];
    auto load_slab = [&](int k0) {
#pragma unroll
        for (int i = 0; i < 4; ++i) {
            int idx = i * 256 + tid;
            int row = idx >> 3, kq = (idx & 7) << 2;
            ar[i] = *(const float4*)(hs + (size_t)(m0 + row) * HID + k0 + kq);
        }
#pragma unroll
        for (int i = 0; i < 2; ++i) {
            int idx = i * 256 + tid;
            int row = idx >> 3, kq = (idx & 7) << 2;
            br[i] = *(const float4*)(W + (size_t)(n0 + row) * HID + k0 + kq);
        }
    };
    auto store_slab = [&]() {
#pragma unroll
        for (int i = 0; i < 4; ++i) {
            int idx = i * 256 + tid;
            int row = idx >> 3, kq = (idx & 7) << 2;
            _Float16* d = &Ah[row * 48 + kq];
            d[0] = (_Float16)ar[i].x; d[1] = (_Float16)ar[i].y;
            d[2] = (_Float16)ar[i].z; d[3] = (_Float16)ar[i].w;
        }
#pragma unroll
        for (int i = 0; i < 2; ++i) {
            int idx = i * 256 + tid;
            int row = idx >> 3, kq = (idx & 7) << 2;
            _Float16* d = &Bh[row * 48 + kq];
            d[0] = (_Float16)br[i].x; d[1] = (_Float16)br[i].y;
            d[2] = (_Float16)br[i].z; d[3] = (_Float16)br[i].w;
        }
    };

    load_slab(0);
    for (int k0 = 0; k0 < HID; k0 += 32) {
        store_slab();
        __syncthreads();
        if (k0 + 32 < HID) load_slab(k0 + 32);   // overlaps with WMMAs below

        v16h a0 = load_frag16(&Ah[(wm * 32 +  0) * 48], 48);
        v16h a1 = load_frag16(&Ah[(wm * 32 + 16) * 48], 48);
        v16h b0 = load_frag16(&Bh[(wn * 32 +  0) * 48], 48);
        v16h b1 = load_frag16(&Bh[(wn * 32 + 16) * 48], 48);
        c00 = wmma_f16(a0, b0, c00); c01 = wmma_f16(a0, b1, c01);
        c10 = wmma_f16(a1, b0, c10); c11 = wmma_f16(a1, b1, c11);
        __syncthreads();
    }

    // Stage accumulators in LDS so RoPE can pair columns d and d+32.
    {
        const int lane = tid & 31;
        const int half = lane >> 4;
        const int nl   = lane & 15;
#pragma unroll
        for (int r = 0; r < 8; ++r) {
            int rm = r + 8 * half;
            Cs[(wm * 32 +  0 + rm) * 64 + wn * 32 +  0 + nl] = c00[r];
            Cs[(wm * 32 +  0 + rm) * 64 + wn * 32 + 16 + nl] = c01[r];
            Cs[(wm * 32 + 16 + rm) * 64 + wn * 32 +  0 + nl] = c10[r];
            Cs[(wm * 32 + 16 + rm) * 64 + wn * 32 + 16 + nl] = c11[r];
        }
    }
    __syncthreads();

    const bool rope = (qkv < 2);
    for (int i = 0; i < 16; ++i) {
        int e   = i * 256 + tid;                // 4096 pairs
        int row = e >> 5;
        int d   = (e & 31) << 1;                // even; d,d+1 same RoPE half
        int sg  = m0 + row;
        float o0, o1;
        float v0 = Cs[row * 64 + d + 0] + bia[n0 + d + 0];
        float v1 = Cs[row * 64 + d + 1] + bia[n0 + d + 1];
        if (rope) {
            int j0 = (d + 0) & 31, j1 = (d + 1) & 31;
            float a0 = (float)sg * __expf(-(float)j0 * (9.210340371976184f / 32.0f));
            float a1 = (float)sg * __expf(-(float)j1 * (9.210340371976184f / 32.0f));
            float s0, c0, s1, c1;
            __sincosf(a0, &s0, &c0);
            __sincosf(a1, &s1, &c1);
            if (d < 32) {
                float p0 = Cs[row * 64 + d + 32] + bia[n0 + d + 32];
                float p1 = Cs[row * 64 + d + 33] + bia[n0 + d + 33];
                o0 = v0 * c0 - p0 * s0;
                o1 = v1 * c1 - p1 * s1;
            } else {
                float p0 = Cs[row * 64 + d - 32] + bia[n0 + d - 32];
                float p1 = Cs[row * 64 + d - 31] + bia[n0 + d - 31];
                o0 = p0 * s0 + v0 * c0;
                o1 = p1 * s1 + v1 * c1;
            }
        } else {
            o0 = v0; o1 = v1;
        }
        v2h pair = { (_Float16)o0, (_Float16)o1 };
        *(v2h*)&dst[((size_t)head * S_LEN + sg) * HD + d] = pair;
    }
}

// ---------------------------------------------------------------------------
// Kernel 2: flash attention with split-phase, double-buffered tile movement.
// 8 waves x 16 query rows = 128 q rows per block.  At the top of iteration i,
// wave 0 issues the TDM for K block i+1 into the other buffer (TENSORcnt
// drained only at iteration end) and all threads issue V block i+1 global
// loads; compute (16 WMMAs/wave + online softmax) runs on buffer i meanwhile.
// ---------------------------------------------------------------------------
__global__ __launch_bounds__(256)
void flash_attn_kernel(const _Float16* __restrict__ Qh,
                       const _Float16* __restrict__ Kh,
                       const _Float16* __restrict__ Vh,
                       const float* __restrict__ mask,
                       float* __restrict__ out)
{
    __shared__ __align__(16) _Float16 Kt[2][64 * 72];   // [key][d], TDM-padded
    __shared__ __align__(16) _Float16 Vt[2][64 * 72];   // [d][key] (transposed)
    __shared__ __align__(16) _Float16 Pt[8 * 16 * 72];  // per-wave P scratch

    const int tid  = threadIdx.x;
    const int wave = tid >> 5;
    const int lane = tid & 31;
    const int half = lane >> 4;
    const int nl   = lane & 15;
    const int head = blockIdx.y;
    const int q0   = blockIdx.x * 128 + wave * 16;

    const _Float16* qbase = Qh + ((size_t)head * S_LEN + q0) * HD;
    v16h qf0 = load_frag16(qbase +  0, HD);
    v16h qf1 = load_frag16(qbase + 32, HD);

    const int vrow = tid >> 3;              // tile loader coords (per thread)
    const int vc8  = (tid & 7) << 3;

    v8f o[4];
    float m_i[8], l_i[8];
#pragma unroll
    for (int j = 0; j < 4; ++j) { v8f z = {}; o[j] = z; }
#pragma unroll
    for (int r = 0; r < 8; ++r) { m_i[r] = -1e30f; l_i[r] = 0.0f; }

    // register prefetch of the V (and, without TDM, K) tile
    v8h vreg[2];
#ifndef HAVE_TDM
    v8h kreg[2];
#endif
    auto load_tile_regs = [&](int kb) {
#pragma unroll
        for (int i = 0; i < 2; ++i) {
            int row = vrow + i * 32;
            vreg[i] = *(const v8h*)(Vh + ((size_t)head * S_LEN + kb + row) * HD + vc8);
#ifndef HAVE_TDM
            kreg[i] = *(const v8h*)(Kh + ((size_t)head * S_LEN + kb + row) * HD + vc8);
#endif
        }
    };
    auto store_tile = [&](int buf) {
#pragma unroll
        for (int i = 0; i < 2; ++i) {
            int row = vrow + i * 32;
#ifndef HAVE_TDM
            *(v8h*)&Kt[buf][row * 72 + vc8] = kreg[i];
#endif
#pragma unroll
            for (int e = 0; e < 8; ++e) Vt[buf][(vc8 + e) * 72 + row] = vreg[i][e];
        }
    };

    // Prologue: land block 0 in buffer 0.
    load_tile_regs(0);
#ifdef HAVE_TDM
    if (wave == 0)
        tdm_load_tile_64x64_f16(Kh + (size_t)head * S_LEN * HD,
                                (unsigned)(uintptr_t)&Kt[0][0]);
#endif
    store_tile(0);
#ifdef HAVE_TDM
    if (wave == 0) __builtin_amdgcn_s_wait_tensorcnt(0);
#endif
    __syncthreads();

    for (int kb = 0; kb < S_LEN; kb += 64) {
        const int  buf      = (kb >> 6) & 1;
        const bool has_next = (kb + 64) < S_LEN;

        // Split-phase prefetch of block i+1 into the other buffer.
        if (has_next) {
#ifdef HAVE_TDM
            if (wave == 0)
                tdm_load_tile_64x64_f16(Kh + ((size_t)head * S_LEN + kb + 64) * HD,
                                        (unsigned)(uintptr_t)&Kt[buf ^ 1][0]);
#endif
            load_tile_regs(kb + 64);
        }

        // S = Q K^T for this wave's 16 rows x 64 keys
        v8f sc[4];
#pragma unroll
        for (int j = 0; j < 4; ++j) {
            v8f z = {}; sc[j] = z;
            v16h bk0 = load_frag16(&Kt[buf][(j * 16) * 72 +  0], 72);
            v16h bk1 = load_frag16(&Kt[buf][(j * 16) * 72 + 32], 72);
            sc[j] = wmma_f16(qf0, bk0, sc[j]);
            sc[j] = wmma_f16(qf1, bk1, sc[j]);
        }

        float mk[4];
#pragma unroll
        for (int j = 0; j < 4; ++j) mk[j] = mask[kb + j * 16 + nl];
#pragma unroll
        for (int j = 0; j < 4; ++j)
#pragma unroll
            for (int r = 0; r < 8; ++r) sc[j][r] = sc[j][r] * 0.125f + mk[j];

        // Online softmax: a row's 16 N values live across lanes 0..15 of a
        // half-wave, so reduce with xor masks 1,2,4,8 only.
        float alpha[8];
#pragma unroll
        for (int r = 0; r < 8; ++r) {
            float v = fmaxf(fmaxf(sc[0][r], sc[1][r]), fmaxf(sc[2][r], sc[3][r]));
#pragma unroll
            for (int off = 1; off <= 8; off <<= 1)
                v = fmaxf(v, __shfl_xor(v, off, 32));
            float mnew = fmaxf(m_i[r], v);
            alpha[r] = __expf(m_i[r] - mnew);
            m_i[r] = mnew;
        }
#pragma unroll
        for (int j = 0; j < 4; ++j)
#pragma unroll
            for (int r = 0; r < 8; ++r) sc[j][r] = __expf(sc[j][r] - m_i[r]);
#pragma unroll
        for (int r = 0; r < 8; ++r) {
            float v = sc[0][r] + sc[1][r] + sc[2][r] + sc[3][r];
#pragma unroll
            for (int off = 1; off <= 8; off <<= 1)
                v += __shfl_xor(v, off, 32);
            l_i[r] = l_i[r] * alpha[r] + v;
        }
#pragma unroll
        for (int j = 0; j < 4; ++j)
#pragma unroll
            for (int r = 0; r < 8; ++r) o[j][r] *= alpha[r];

        // P: C layout -> A layout via per-wave LDS round trip (f16).
        _Float16* pw = &Pt[wave * 16 * 72];
#pragma unroll
        for (int j = 0; j < 4; ++j)
#pragma unroll
            for (int r = 0; r < 8; ++r)
                pw[(r + 8 * half) * 72 + j * 16 + nl] = (_Float16)sc[j][r];

        v16h pa0 = load_frag16(pw +  0, 72);
        v16h pa1 = load_frag16(pw + 32, 72);
#pragma unroll
        for (int j = 0; j < 4; ++j) {
            v16h bv0 = load_frag16(&Vt[buf][(j * 16) * 72 +  0], 72);
            v16h bv1 = load_frag16(&Vt[buf][(j * 16) * 72 + 32], 72);
            o[j] = wmma_f16(pa0, bv0, o[j]);
            o[j] = wmma_f16(pa1, bv1, o[j]);
        }

        // Land the prefetched V tile; drain the TDM before the barrier so
        // everyone can read Kt[buf^1] next iteration.
        if (has_next) {
            store_tile(buf ^ 1);
#ifdef HAVE_TDM
            if (wave == 0) __builtin_amdgcn_s_wait_tensorcnt(0);
#endif
        }
        __syncthreads();
    }

    // Normalize and write f32 context back to [s][h*64+d]; write-once stream,
    // so use non-temporal stores to keep it out of the caches.
#pragma unroll
    for (int r = 0; r < 8; ++r) {
        float inv = 1.0f / l_i[r];
        int srow = q0 + r + 8 * half;
#pragma unroll
        for (int j = 0; j < 4; ++j)
            __builtin_nontemporal_store(o[j][r] * inv,
                &out[(size_t)srow * HID + head * HD + j * 16 + nl]);
    }
}

extern "C" void kernel_launch(void* const* d_in, const int* in_sizes, int n_in,
                              void* d_out, int out_size, void* d_ws, size_t ws_size,
                              hipStream_t stream) {
    const float* hs   = (const float*)d_in[0];
    const float* mask = (const float*)d_in[1];
    const float* Wq   = (const float*)d_in[2];
    const float* bq   = (const float*)d_in[3];
    const float* Wk   = (const float*)d_in[4];
    const float* bk   = (const float*)d_in[5];
    const float* Wv   = (const float*)d_in[6];
    const float* bv   = (const float*)d_in[7];
    float* out = (float*)d_out;

    // f16 head-major staging: Q, K, V each NH*S*HD halfs (6 MB) in workspace.
    _Float16* Qh = (_Float16*)d_ws;
    const size_t per = (size_t)NH * S_LEN * HD;
    _Float16* Kh = Qh + per;
    _Float16* Vh = Kh + per;

    dim3 g1(S_LEN / 128, NH, 3);
    qkv_rope_kernel<<<g1, 256, 0, stream>>>(hs, Wq, Wk, Wv, bq, bk, bv, Qh, Kh, Vh);

    dim3 g2(S_LEN / 128, NH);
    flash_attn_kernel<<<g2, 256, 0, stream>>>(Qh, Kh, Vh, mask, out);
}
